// GRU_64854006169634
// MI455X (gfx1250) — matrix operand order repaired and checked by reading, not compiled, not run on hardware
//
#include <hip/hip_runtime.h>
#include <hip/hip_bf16.h>
#include <math.h>

// Problem constants (match reference)
#define B_   256
#define T_   256
#define D_   512
#define H_   1024
#define C_   1000
#define CP_  1024   // C padded to multiple of 16 for WMMA N-dim
#define M1_  (B_ * T_)   // 65536 rows for the input-projection GEMM

typedef __bf16 bf16;
typedef __attribute__((ext_vector_type(16))) __bf16 v16bf;
typedef __attribute__((ext_vector_type(8)))  __bf16 v8bf;
typedef __attribute__((ext_vector_type(8)))  float  v8f;

// ---------------------------------------------------------------------------
// WMMA helpers (CDNA5: D = A(16x32 bf16) x B(32x16 bf16) + C(16x16 f32))
// ---------------------------------------------------------------------------
__device__ __forceinline__ v8f wmma_bf16(v16bf a, v16bf b, v8f c) {
    // (neg_a, A, neg_b, B, c_mod, C, reuse_a, reuse_b)
    return __builtin_amdgcn_wmma_f32_16x16x32_bf16(false, a, false, b,
                                                   (short)0, c, false, false);
}

// A fragment: 16x32 tile of row-major A[M x K] (lda = K).
// ISA layout: lanes 0-15 hold row M=l with K {0..7,16..23}; lanes 16-31 hold
// row M=l with K {8..15,24..31}.
__device__ __forceinline__ v16bf load_frag_a(const bf16* __restrict__ A,
                                             int lda, int row0, int k0, int lane) {
    const int half = lane >> 4;
    const int l    = lane & 15;
    const bf16* p = A + (size_t)(row0 + l) * lda + k0 + half * 8;
    v8bf lo = *(const v8bf*)(p);        // K = half*8 + 0..7
    v8bf hi = *(const v8bf*)(p + 16);   // K = half*8 + 16..23
    v16bf r;
#pragma unroll
    for (int i = 0; i < 8; ++i) { r[i] = lo[i]; r[8 + i] = hi[i]; }
    return r;
}

// B fragment: 32x16 tile. B is supplied pre-transposed: BT[N x K] row-major
// (ldb = K), so column n of B is row n of BT (contiguous in K).
// ISA layout: lanes 0-15 hold col N=l, K {0..7,16..23}; lanes 16-31 hold col N=l, K {8..15,24..31}.
__device__ __forceinline__ v16bf load_frag_b(const bf16* __restrict__ BT,
                                             int ldb, int col0, int k0, int lane) {
    const int half = lane >> 4;
    const int l    = lane & 15;
    const bf16* p = BT + (size_t)(col0 + l) * ldb + k0 + half * 8;
    v8bf lo = *(const v8bf*)(p);
    v8bf hi = *(const v8bf*)(p + 16);
    v16bf r;
#pragma unroll
    for (int i = 0; i < 8; ++i) { r[i] = lo[i]; r[8 + i] = hi[i]; }
    return r;
}

__device__ __forceinline__ float sigmoidf_(float x) {
    return 1.0f / (1.0f + expf(-x));
}

// ---------------------------------------------------------------------------
// Utility kernels
// ---------------------------------------------------------------------------
__global__ void cvt_f32_to_bf16_kernel(const float* __restrict__ src,
                                       bf16* __restrict__ dst, size_t n) {
    size_t i = (size_t)blockIdx.x * blockDim.x + threadIdx.x;
    size_t stride = (size_t)gridDim.x * blockDim.x;
    for (; i < n; i += stride) dst[i] = (bf16)src[i];
}

// dst[c*R + r] = (c < Cc) ? src[r*Cc + c] : 0   for c in [0,Cpad), r in [0,R)
__global__ void transpose_to_bf16_kernel(const float* __restrict__ src,
                                         bf16* __restrict__ dst,
                                         int R, int Cc, int Cpad) {
    size_t i = (size_t)blockIdx.x * blockDim.x + threadIdx.x;
    size_t n = (size_t)Cpad * R;
    if (i >= n) return;
    int c = (int)(i / R);
    int r = (int)(i % R);
    dst[i] = (c < Cc) ? (bf16)src[(size_t)r * Cc + c] : (bf16)0.0f;
}

__global__ void zero_h_kernel(float* __restrict__ h, bf16* __restrict__ hbf, int n) {
    int i = blockIdx.x * blockDim.x + threadIdx.x;
    if (i < n) { h[i] = 0.0f; hbf[i] = (bf16)0.0f; }
}

// ---------------------------------------------------------------------------
// Phase 1: xz/xr/xh = x @ {W_z, W_r, W}  fused into one pass over A.
// A: bf16 [M x K], BT*: bf16 [N x K], O*: f32 [M x N].
// block = 128 threads (4 wave32). Wave tile = 32x32 (2 M-frags x 2 N-frags),
// block tile = 64x64. Per K-chunk: 16 b128 loads feed 12 WMMAs.
// ---------------------------------------------------------------------------
__global__ void __launch_bounds__(128)
gemm3_kernel(const bf16* __restrict__ A,
             const bf16* __restrict__ BT0, const bf16* __restrict__ BT1,
             const bf16* __restrict__ BT2,
             float* __restrict__ O0, float* __restrict__ O1,
             float* __restrict__ O2,
             int M, int N, int K) {
    const int wave = threadIdx.x >> 5;
    const int lane = threadIdx.x & 31;
    const int m0 = blockIdx.x * 64 + (wave >> 1) * 32;
    const int n0 = blockIdx.y * 64 + (wave & 1) * 32;

    v8f acc[3][2][2];
#pragma unroll
    for (int g = 0; g < 3; ++g)
#pragma unroll
        for (int mi = 0; mi < 2; ++mi)
#pragma unroll
            for (int ni = 0; ni < 2; ++ni) acc[g][mi][ni] = (v8f){};

    const bf16* BT[3] = {BT0, BT1, BT2};
    for (int k = 0; k < K; k += 32) {
        v16bf a0 = load_frag_a(A, K, m0,      k, lane);
        v16bf a1 = load_frag_a(A, K, m0 + 16, k, lane);
#pragma unroll
        for (int g = 0; g < 3; ++g) {
            v16bf b0 = load_frag_b(BT[g], K, n0,      k, lane);
            v16bf b1 = load_frag_b(BT[g], K, n0 + 16, k, lane);
            acc[g][0][0] = wmma_bf16(a0, b0, acc[g][0][0]);
            acc[g][0][1] = wmma_bf16(a0, b1, acc[g][0][1]);
            acc[g][1][0] = wmma_bf16(a1, b0, acc[g][1][0]);
            acc[g][1][1] = wmma_bf16(a1, b1, acc[g][1][1]);
        }
    }
    const int half = lane >> 4;
    const int l    = lane & 15;
    float* O[3] = {O0, O1, O2};
#pragma unroll
    for (int mi = 0; mi < 2; ++mi)
#pragma unroll
        for (int ni = 0; ni < 2; ++ni) {
            const int col = n0 + ni * 16 + l;
#pragma unroll
            for (int j = 0; j < 8; ++j) {
                size_t idx = (size_t)(m0 + mi * 16 + half * 8 + j) * N + col;
#pragma unroll
                for (int g = 0; g < 3; ++g) O[g][idx] = acc[g][mi][ni][j];
            }
        }
}

// ---------------------------------------------------------------------------
// Phase 2: one fused GRU timestep.
//   z = sigmoid(xz_t + h Uz); r = sigmoid(xr_t + h Ur)
//   h_hat = tanh(xh_t + (h U) * r); h' = z*h + (1-z)*h_hat
// A = h (bf16 [B x H]); UT* = bf16 [H x H] (pre-transposed).
// xz/xr/xh stored as [B*T, H] with row index b*T + t.
// Wave tile 32x32 x 3 gates; block tile 64x64; grid (4,16).
// ---------------------------------------------------------------------------
__global__ void __launch_bounds__(128)
gru_step_kernel(const bf16*  __restrict__ hbf_in,
                const float* __restrict__ h_in,
                const bf16*  __restrict__ UzT,
                const bf16*  __restrict__ UrT,
                const bf16*  __restrict__ UT,
                const float* __restrict__ xz,
                const float* __restrict__ xr,
                const float* __restrict__ xh,
                int t,
                float* __restrict__ h_out,
                bf16*  __restrict__ hbf_out) {
    const int wave = threadIdx.x >> 5;
    const int lane = threadIdx.x & 31;
    const int m0 = blockIdx.x * 64 + (wave >> 1) * 32;   // batch rows
    const int n0 = blockIdx.y * 64 + (wave & 1) * 32;    // hidden cols

    v8f acc[3][2][2];
#pragma unroll
    for (int g = 0; g < 3; ++g)
#pragma unroll
        for (int mi = 0; mi < 2; ++mi)
#pragma unroll
            for (int ni = 0; ni < 2; ++ni) acc[g][mi][ni] = (v8f){};

    const bf16* UTg[3] = {UzT, UrT, UT};
    for (int k = 0; k < H_; k += 32) {
        v16bf a0 = load_frag_a(hbf_in, H_, m0,      k, lane);
        v16bf a1 = load_frag_a(hbf_in, H_, m0 + 16, k, lane);
#pragma unroll
        for (int g = 0; g < 3; ++g) {
            v16bf b0 = load_frag_b(UTg[g], H_, n0,      k, lane);
            v16bf b1 = load_frag_b(UTg[g], H_, n0 + 16, k, lane);
            acc[g][0][0] = wmma_bf16(a0, b0, acc[g][0][0]);
            acc[g][0][1] = wmma_bf16(a0, b1, acc[g][0][1]);
            acc[g][1][0] = wmma_bf16(a1, b0, acc[g][1][0]);
            acc[g][1][1] = wmma_bf16(a1, b1, acc[g][1][1]);
        }
    }

    const int half = lane >> 4;
    const int l    = lane & 15;
#pragma unroll
    for (int mi = 0; mi < 2; ++mi)
#pragma unroll
        for (int ni = 0; ni < 2; ++ni) {
            const int col = n0 + ni * 16 + l;
#pragma unroll
            for (int j = 0; j < 8; ++j) {
                const int b = m0 + mi * 16 + half * 8 + j;   // batch index
                const size_t hidx = (size_t)b * H_ + col;
                const size_t xidx = ((size_t)b * T_ + t) * H_ + col;
                float z  = sigmoidf_(xz[xidx] + acc[0][mi][ni][j]);
                float r  = sigmoidf_(xr[xidx] + acc[1][mi][ni][j]);
                float hh = tanhf(xh[xidx] + acc[2][mi][ni][j] * r); // reset AFTER matmul
                float hn = z * h_in[hidx] + (1.0f - z) * hh;
                h_out[hidx]   = hn;
                hbf_out[hidx] = (bf16)hn;
            }
        }
}

// ---------------------------------------------------------------------------
// Phase 3a: logits = h_last @ W_ph (padded to CP_ cols), f32 out.
// Wave tile 32x32, block tile 64x64.
// ---------------------------------------------------------------------------
__global__ void __launch_bounds__(128)
gemm1_kernel(const bf16* __restrict__ A, const bf16* __restrict__ BT,
             float* __restrict__ O, int M, int N, int K) {
    const int wave = threadIdx.x >> 5;
    const int lane = threadIdx.x & 31;
    const int m0 = blockIdx.x * 64 + (wave >> 1) * 32;
    const int n0 = blockIdx.y * 64 + (wave & 1) * 32;

    v8f acc[2][2];
#pragma unroll
    for (int mi = 0; mi < 2; ++mi)
#pragma unroll
        for (int ni = 0; ni < 2; ++ni) acc[mi][ni] = (v8f){};

    for (int k = 0; k < K; k += 32) {
        v16bf a0 = load_frag_a(A, K, m0,      k, lane);
        v16bf a1 = load_frag_a(A, K, m0 + 16, k, lane);
        v16bf b0 = load_frag_b(BT, K, n0,      k, lane);
        v16bf b1 = load_frag_b(BT, K, n0 + 16, k, lane);
        acc[0][0] = wmma_bf16(a0, b0, acc[0][0]);
        acc[0][1] = wmma_bf16(a0, b1, acc[0][1]);
        acc[1][0] = wmma_bf16(a1, b0, acc[1][0]);
        acc[1][1] = wmma_bf16(a1, b1, acc[1][1]);
    }
    const int half = lane >> 4;
    const int l    = lane & 15;
#pragma unroll
    for (int mi = 0; mi < 2; ++mi)
#pragma unroll
        for (int ni = 0; ni < 2; ++ni) {
            const int col = n0 + ni * 16 + l;
#pragma unroll
            for (int j = 0; j < 8; ++j)
                O[(size_t)(m0 + mi * 16 + half * 8 + j) * N + col] = acc[mi][ni][j];
        }
}

// Phase 3b: out[b, :] = log_softmax(logits[b, :C] + bias)
__global__ void __launch_bounds__(256)
logsoftmax_kernel(const float* __restrict__ logits, const float* __restrict__ bias,
                  float* __restrict__ out) {
    __shared__ float red[256];
    const int b   = blockIdx.x;
    const int tid = threadIdx.x;

    float mx = -INFINITY;
    for (int c = tid; c < C_; c += 256)
        mx = fmaxf(mx, logits[(size_t)b * CP_ + c] + bias[c]);
    red[tid] = mx; __syncthreads();
    for (int s = 128; s > 0; s >>= 1) {
        if (tid < s) red[tid] = fmaxf(red[tid], red[tid + s]);
        __syncthreads();
    }
    mx = red[0]; __syncthreads();

    float acc = 0.0f;
    for (int c = tid; c < C_; c += 256)
        acc += expf(logits[(size_t)b * CP_ + c] + bias[c] - mx);
    red[tid] = acc; __syncthreads();
    for (int s = 128; s > 0; s >>= 1) {
        if (tid < s) red[tid] += red[tid + s];
        __syncthreads();
    }
    const float lse = logf(red[0]);

    for (int c = tid; c < C_; c += 256)
        out[(size_t)b * C_ + c] = logits[(size_t)b * CP_ + c] + bias[c] - mx - lse;
}

// ---------------------------------------------------------------------------
// Host-side orchestration
// ---------------------------------------------------------------------------
extern "C" void kernel_launch(void* const* d_in, const int* in_sizes, int n_in,
                              void* d_out, int out_size, void* d_ws, size_t ws_size,
                              hipStream_t stream) {
    const float* x    = (const float*)d_in[0];   // [B,T,D]
    const float* W_z  = (const float*)d_in[1];   // [D,H]
    const float* W_r  = (const float*)d_in[2];   // [D,H]
    const float* W    = (const float*)d_in[3];   // [D,H]
    const float* U_z  = (const float*)d_in[4];   // [H,H]
    const float* U_r  = (const float*)d_in[5];   // [H,H]
    const float* U    = (const float*)d_in[6];   // [H,H]
    const float* W_ph = (const float*)d_in[7];   // [H,C]
    const float* b_p  = (const float*)d_in[8];   // [C]
    float* out = (float*)d_out;                  // [B,C] log-probs

    // ---- workspace carve (all 256B aligned); total ~890 MB ----
    char* w = (char*)d_ws;
    auto carve = [&](size_t bytes) {
        void* p = (void*)w;
        w += (bytes + 255) & ~(size_t)255;
        return p;
    };
    bf16*  xbf   = (bf16*) carve((size_t)M1_ * D_ * sizeof(bf16));
    bf16*  WzT   = (bf16*) carve((size_t)H_  * D_ * sizeof(bf16));
    bf16*  WrT   = (bf16*) carve((size_t)H_  * D_ * sizeof(bf16));
    bf16*  WT    = (bf16*) carve((size_t)H_  * D_ * sizeof(bf16));
    bf16*  UzT   = (bf16*) carve((size_t)H_  * H_ * sizeof(bf16));
    bf16*  UrT   = (bf16*) carve((size_t)H_  * H_ * sizeof(bf16));
    bf16*  UT    = (bf16*) carve((size_t)H_  * H_ * sizeof(bf16));
    bf16*  WphT  = (bf16*) carve((size_t)CP_ * H_ * sizeof(bf16));
    float* xz    = (float*)carve((size_t)M1_ * H_ * sizeof(float));
    float* xr    = (float*)carve((size_t)M1_ * H_ * sizeof(float));
    float* xh    = (float*)carve((size_t)M1_ * H_ * sizeof(float));
    float* hA    = (float*)carve((size_t)B_  * H_ * sizeof(float));
    float* hB    = (float*)carve((size_t)B_  * H_ * sizeof(float));
    bf16*  hAbf  = (bf16*) carve((size_t)B_  * H_ * sizeof(bf16));
    bf16*  hBbf  = (bf16*) carve((size_t)B_  * H_ * sizeof(bf16));
    float* logit = (float*)carve((size_t)B_  * CP_ * sizeof(float));

    // ---- convert / transpose weights and inputs to bf16 ----
    {
        size_t n = (size_t)M1_ * D_;
        cvt_f32_to_bf16_kernel<<<4096, 256, 0, stream>>>(x, xbf, n);
    }
    {
        size_t nW = (size_t)H_ * D_;   // dst elements for each W transpose
        int blocks = (int)((nW + 255) / 256);
        transpose_to_bf16_kernel<<<blocks, 256, 0, stream>>>(W_z, WzT, D_, H_, H_);
        transpose_to_bf16_kernel<<<blocks, 256, 0, stream>>>(W_r, WrT, D_, H_, H_);
        transpose_to_bf16_kernel<<<blocks, 256, 0, stream>>>(W,   WT,  D_, H_, H_);
        size_t nU = (size_t)H_ * H_;
        int blocksU = (int)((nU + 255) / 256);
        transpose_to_bf16_kernel<<<blocksU, 256, 0, stream>>>(U_z, UzT, H_, H_, H_);
        transpose_to_bf16_kernel<<<blocksU, 256, 0, stream>>>(U_r, UrT, H_, H_, H_);
        transpose_to_bf16_kernel<<<blocksU, 256, 0, stream>>>(U,   UT,  H_, H_, H_);
        size_t nP = (size_t)CP_ * H_;
        int blocksP = (int)((nP + 255) / 256);
        transpose_to_bf16_kernel<<<blocksP, 256, 0, stream>>>(W_ph, WphT, H_, C_, CP_);
    }

    // ---- phase 1: input projections, 3 gates in one pass ----
    {
        dim3 grid(M1_ / 64, H_ / 64);
        gemm3_kernel<<<grid, 128, 0, stream>>>(xbf, WzT, WrT, WT,
                                               xz, xr, xh, M1_, H_, D_);
    }

    // ---- h0 = 0 ----
    zero_h_kernel<<<(B_ * H_ + 255) / 256, 256, 0, stream>>>(hA, hAbf, B_ * H_);

    // ---- phase 2: 256 sequential fused timesteps (ping-pong h) ----
    {
        dim3 grid(B_ / 64, H_ / 64);
        int cur = 0;
        for (int t = 0; t < T_; ++t) {
            const bf16*  hin_bf = (cur == 0) ? hAbf : hBbf;
            const float* hin    = (cur == 0) ? hA   : hB;
            float*       hout   = (cur == 0) ? hB   : hA;
            bf16*        houtbf = (cur == 0) ? hBbf : hAbf;
            gru_step_kernel<<<grid, 128, 0, stream>>>(hin_bf, hin, UzT, UrT, UT,
                                                      xz, xr, xh, t, hout, houtbf);
            cur ^= 1;
        }
        // final hidden state (bf16) lives in the buffer indexed by `cur`
        const bf16* hlast_bf = (cur == 0) ? hAbf : hBbf;

        // ---- phase 3: projection + log_softmax ----
        dim3 gridP(B_ / 64, CP_ / 64);
        gemm1_kernel<<<gridP, 128, 0, stream>>>(hlast_bf, WphT, logit, B_, CP_, H_);
        logsoftmax_kernel<<<B_, 256, 0, stream>>>(logit, b_p, out);
    }
}
